// LLPAttentionModel_22617297780973
// MI455X (gfx1250) — compile-verified, matrix operand order checked by reading, not run on hardware
//
#include <hip/hip_runtime.h>
#include <hip/hip_bf16.h>
#include <stdint.h>
#include <math.h>

// ---------------------------------------------------------------------------
// Types for CDNA5 WMMA (wave32, 16x16x32 bf16 -> f32)
// ---------------------------------------------------------------------------
typedef __bf16 bf16_t;
typedef __attribute__((ext_vector_type(16))) __bf16 v16bf;
typedef __attribute__((ext_vector_type(8)))  float  v8f;
typedef __attribute__((ext_vector_type(4)))  unsigned int v4u;

union FragU { v16bf v; v4u q[2]; };

__device__ __forceinline__ v8f wmma_bf16(v16bf a, v16bf b, v8f c) {
    // D = A(16x32) * B(32x16) + C ; args: neg_a, A, neg_b, B, c_mod, C, reuse_a, reuse_b
    return __builtin_amdgcn_wmma_f32_16x16x32_bf16(false, a, false, b, (short)0, c, false, false);
}

// A-type fragment loader (also used for B^T / V^T / P fragments).
// ISA 16-bit A layout: lane l: row = l%16, h = l/16; elements 0..7 = K[h*8 .. h*8+7],
// elements 8..15 = K[16+h*8 .. 16+h*8+7]  -> two contiguous 16B segments.
__device__ __forceinline__ v16bf lds_frag(const bf16_t* base, int stride, int row0, int k0) {
    int lane = threadIdx.x & 31;
    int r = lane & 15, h = lane >> 4;
    const bf16_t* p = base + (size_t)(row0 + r) * stride + k0 + h * 8;
    FragU f;
    f.q[0] = *(const v4u*)(p);
    f.q[1] = *(const v4u*)(p + 16);
    return f.v;
}

// ---------------------------------------------------------------------------
// Generic bf16 WMMA GEMM: C[M,N] = act(A[M,K] * Bt^T + bias [+ Cprev]) [+ resid]
// Bt is B stored TRANSPOSED: Bt[n*K + k]  (weights are converted into this
// layout once per layer, so no in-kernel transpose scatter is needed).
// Block = 256 thr (8 waves), tile 128(M) x 256(N), K-step 32, double-buffered
// LDS. Wave grid 2(M) x 4(N); wave tile 64x64 = 4x4 WMMA accumulators
// (16 WMMAs per 8 fragment loads per K-step, one barrier per K-step).
// All epilogue features are compile-time to keep the accumulator-live tail
// branch-free (avoids scratch spills of the 128 accumulator VGPRs).
// AROW: A element (m,k) at A[arow[m] + k]   (implicit-conv taps)
// ---------------------------------------------------------------------------
template <bool AROW, bool BIAS, bool CPREV, bool RESID, int ACT, bool OBF16>
__global__ __launch_bounds__(256) void k_gemm(
    const bf16_t* __restrict__ A, int lda, const int* __restrict__ arow,
    const bf16_t* __restrict__ Bt,
    int M, int N, int K,
    const float* __restrict__ bias,   // len N
    const float* __restrict__ cprev,  // MxN fp32 accumulate input
    const float* __restrict__ resid,  // MxN fp32 residual
    void* __restrict__ Cout, int ldc)
{
    __shared__ bf16_t As[2][128 * 40];   // [m][k], k padded 32->40
    __shared__ bf16_t Bs[2][256 * 40];   // [n][k]

    const int tid  = threadIdx.x;
    const int lane = tid & 31;
    const int wave = tid >> 5;
    const int wm = wave >> 2;         // 0..1 -> rows wm*64
    const int wn = wave & 3;          // 0..3 -> cols wn*64
    const int m0 = blockIdx.x * 128;
    const int n0 = blockIdx.y * 256;

    v8f acc[4][4];
#pragma unroll
    for (int i = 0; i < 4; ++i)
#pragma unroll
        for (int j = 0; j < 4; ++j)
            acc[i][j] = (v8f){0.f,0.f,0.f,0.f,0.f,0.f,0.f,0.f};

    const int T = K >> 5;   // K is always a multiple of 32
    v4u ar[2], br[4];

    // ---- tile loaders: global -> regs, regs -> LDS (all b128)
    auto load_tile = [&](int k0) {
#pragma unroll
        for (int i = 0; i < 2; ++i) {
            int id  = tid * 2 + i;          // 0..511
            int row = id >> 2;              // 0..127
            int seg = (id & 3) * 8;         // k offset
            int gr  = m0 + row;
            if (gr < M) {
                const bf16_t* src = AROW ? (A + (size_t)arow[gr] + k0 + seg)
                                         : (A + (size_t)gr * lda + k0 + seg);
                ar[i] = *(const v4u*)src;
            } else {
                ar[i] = (v4u){0u, 0u, 0u, 0u};
            }
        }
#pragma unroll
        for (int i = 0; i < 4; ++i) {
            int id  = tid * 4 + i;          // 0..1023
            int row = id >> 2;              // 0..255
            int seg = (id & 3) * 8;
            br[i] = *(const v4u*)(Bt + (size_t)(n0 + row) * K + k0 + seg);
        }
    };
    auto store_tile = [&](int buf) {
#pragma unroll
        for (int i = 0; i < 2; ++i) {
            int id = tid * 2 + i;
            *(v4u*)&As[buf][(id >> 2) * 40 + (id & 3) * 8] = ar[i];
        }
#pragma unroll
        for (int i = 0; i < 4; ++i) {
            int id = tid * 4 + i;
            *(v4u*)&Bs[buf][(id >> 2) * 40 + (id & 3) * 8] = br[i];
        }
    };

    // ---- software pipeline: one barrier per K-step
    load_tile(0);
    store_tile(0);
    __syncthreads();

    for (int t = 0; t < T; ++t) {
        int cur = t & 1;
        if (t + 1 < T) {
            load_tile((t + 1) << 5);          // overlap with WMMA on `cur`
            if (t + 2 < T)
                __builtin_prefetch(Bt + (size_t)(n0 + (tid & 255)) * K + ((t + 2) << 5), 0, 1);
        }
        v16bf af[4];
#pragma unroll
        for (int mf = 0; mf < 4; ++mf)
            af[mf] = lds_frag(As[cur], 40, wm * 64 + mf * 16, 0);
#pragma unroll
        for (int nf = 0; nf < 4; ++nf) {
            v16bf bf = lds_frag(Bs[cur], 40, wn * 64 + nf * 16, 0);
#pragma unroll
            for (int mf = 0; mf < 4; ++mf)
                acc[mf][nf] = wmma_bf16(af[mf], bf, acc[mf][nf]);
        }
        if (t + 1 < T) {
            store_tile(cur ^ 1);
            __syncthreads();
        }
    }

    // ---- epilogue (branch-free per instantiation).
    // C layout: element e -> row = e + 8*(lane/16), col = lane%16
    const int hh = lane >> 4;
    const int ln = lane & 15;
#pragma unroll
    for (int mf = 0; mf < 4; ++mf)
#pragma unroll
        for (int nf = 0; nf < 4; ++nf) {
            int col = n0 + wn * 64 + nf * 16 + ln;
            float bval = BIAS ? bias[col] : 0.0f;
#pragma unroll
            for (int e = 0; e < 8; ++e) {
                int row = m0 + wm * 64 + mf * 16 + hh * 8 + e;
                if (row >= M) continue;
                float v = acc[mf][nf][e] + bval;
                if (CPREV) v += cprev[(size_t)row * ldc + col];
                if (ACT == 1)      v = 0.5f * v * (1.0f + erff(v * 0.70710678118f));
                else if (ACT == 2) v = fmaxf(v, 0.0f);
                if (RESID) v += resid[(size_t)row * ldc + col];
                if (OBF16) ((bf16_t*)Cout)[(size_t)row * ldc + col] = (bf16_t)v;
                else       ((float*)Cout)[(size_t)row * ldc + col]  = v;
            }
        }
}

// ---------------------------------------------------------------------------
// Flash attention (online softmax), bf16 in/out, fp32 accum.
// qkv rows: seq*seq_rows + token, 2304 cols = [Q(768)|K(768)|V(768)], head-major.
// Block = 8 waves x 16 queries = 128 queries; K/V staged in LDS per 32-key chunk.
// grid: (ceil(N/128), heads=12, seqs)
// ---------------------------------------------------------------------------
__global__ __launch_bounds__(256) void k_flash(
    const bf16_t* __restrict__ qkv, int seq_rows, int N,
    bf16_t* __restrict__ out, float scale)
{
    __shared__ bf16_t Ks[32 * 72];        // [key][dim], dim padded 64->72
    __shared__ bf16_t Vt[64 * 40];        // V^T: [dim][key], key padded 32->40
    __shared__ bf16_t Ps[8 * 16 * 40];    // per-wave P tile [q][key]

    const int tid = threadIdx.x, lane = tid & 31, wave = tid >> 5;
    const int head = blockIdx.y, seq = blockIdx.z;
    const size_t base = (size_t)seq * seq_rows;
    const int hoff = head * 64;
    const int q0 = blockIdx.x * 128 + wave * 16;
    const int r = lane & 15, hh = lane >> 4;

    // Q fragments (K-dim 64 -> two 16x32 frags), row-clamped loads
    int qrow = q0 + r; if (qrow > N - 1) qrow = N - 1;
    const bf16_t* qp = qkv + (base + qrow) * 2304 + hoff;
    FragU qa0, qa1;
    qa0.q[0] = *(const v4u*)(qp + hh * 8);
    qa0.q[1] = *(const v4u*)(qp + 16 + hh * 8);
    qa1.q[0] = *(const v4u*)(qp + 32 + hh * 8);
    qa1.q[1] = *(const v4u*)(qp + 48 + hh * 8);

    v8f o[4];
#pragma unroll
    for (int j = 0; j < 4; ++j) o[j] = (v8f){0.f,0.f,0.f,0.f,0.f,0.f,0.f,0.f};
    float mrow[8], lrow[8];
#pragma unroll
    for (int e = 0; e < 8; ++e) { mrow[e] = -3.0e38f; lrow[e] = 0.0f; }

    const int nchunks = (N + 31) >> 5;
    for (int kc = 0; kc < nchunks; ++kc) {
        // cooperative K/V chunk load (32 keys x 64 dims each)
        {
            int key = tid >> 3;
            int seg = (tid & 7) * 8;
            int krow = kc * 32 + key; if (krow > N - 1) krow = N - 1;
            const bf16_t* kp = qkv + (base + krow) * 2304 + 768 + hoff + seg;
            *(v4u*)&Ks[key * 72 + seg] = *(const v4u*)kp;
            v4u vv = *(const v4u*)(qkv + (base + krow) * 2304 + 1536 + hoff + seg);
            const bf16_t* ve = (const bf16_t*)&vv;
#pragma unroll
            for (int j = 0; j < 8; ++j)
                Vt[(seg + j) * 40 + key] = ve[j];
        }
        __syncthreads();

#pragma unroll
        for (int sub = 0; sub < 2; ++sub) {
            v16bf kf0 = lds_frag(Ks, 72, sub * 16, 0);
            v16bf kf1 = lds_frag(Ks, 72, sub * 16, 32);
            v8f s = (v8f){0.f,0.f,0.f,0.f,0.f,0.f,0.f,0.f};
            s = wmma_bf16(qa0.v, kf0, s);
            s = wmma_bf16(qa1.v, kf1, s);
            bool colvalid = (kc * 32 + sub * 16 + r) < N;
#pragma unroll
            for (int e = 0; e < 8; ++e) {
                float sv = colvalid ? s[e] * scale : -3.0e38f;
                float x = sv;
                x = fmaxf(x, __shfl_xor(x, 1));
                x = fmaxf(x, __shfl_xor(x, 2));
                x = fmaxf(x, __shfl_xor(x, 4));
                x = fmaxf(x, __shfl_xor(x, 8));
                float mnew  = fmaxf(mrow[e], x);
                float alpha = __expf(mrow[e] - mnew);
                float p     = __expf(sv - mnew);
                float psum = p;
                psum += __shfl_xor(psum, 1);
                psum += __shfl_xor(psum, 2);
                psum += __shfl_xor(psum, 4);
                psum += __shfl_xor(psum, 8);
                lrow[e] = lrow[e] * alpha + psum;
                mrow[e] = mnew;
                o[0][e] *= alpha; o[1][e] *= alpha; o[2][e] *= alpha; o[3][e] *= alpha;
                Ps[(wave * 16 + hh * 8 + e) * 40 + sub * 16 + r] = (bf16_t)p;
            }
        }
        // O += P(16x32) * V(32x64)
        v16bf pf = lds_frag(&Ps[wave * 16 * 40], 40, 0, 0);
#pragma unroll
        for (int j = 0; j < 4; ++j) {
            v16bf vf = lds_frag(Vt, 40, j * 16, 0);
            o[j] = wmma_bf16(pf, vf, o[j]);
        }
        __syncthreads();
    }

#pragma unroll
    for (int e = 0; e < 8; ++e) {
        int row = q0 + hh * 8 + e;
        if (row < N) {
            float inv = 1.0f / lrow[e];
#pragma unroll
            for (int j = 0; j < 4; ++j)
                out[(base + row) * 768 + hoff + j * 16 + r] = (bf16_t)(o[j][e] * inv);
        }
    }
}

// ---------------------------------------------------------------------------
// LayerNorm: one 256-thread block per row of 768. mode 0: src row = out row.
// mode 1 (local tokens): src row = (r/1024)*1025 + 1 + (r%1024).
// ---------------------------------------------------------------------------
__global__ __launch_bounds__(256) void k_layernorm(
    const float* __restrict__ X, const float* __restrict__ w, const float* __restrict__ b,
    void* __restrict__ Y, int out_bf16, int mode)
{
    __shared__ float red[256];
    int rowo = blockIdx.x;
    int rows = (mode == 1) ? ((rowo >> 10) * 1025 + 1 + (rowo & 1023)) : rowo;
    const float* xr = X + (size_t)rows * 768;
    int t = threadIdx.x;
    float x0 = xr[t], x1 = xr[t + 256], x2 = xr[t + 512];
    red[t] = x0 + x1 + x2; __syncthreads();
    for (int s2 = 128; s2 > 0; s2 >>= 1) { if (t < s2) red[t] += red[t + s2]; __syncthreads(); }
    float mean = red[0] * (1.0f / 768.0f);
    __syncthreads();
    float d0 = x0 - mean, d1 = x1 - mean, d2 = x2 - mean;
    red[t] = d0 * d0 + d1 * d1 + d2 * d2; __syncthreads();
    for (int s2 = 128; s2 > 0; s2 >>= 1) { if (t < s2) red[t] += red[t + s2]; __syncthreads(); }
    float rstd = rsqrtf(red[0] * (1.0f / 768.0f) + 1e-5f);
    float y0 = d0 * rstd * w[t]       + b[t];
    float y1 = d1 * rstd * w[t + 256] + b[t + 256];
    float y2 = d2 * rstd * w[t + 512] + b[t + 512];
    if (out_bf16) {
        bf16_t* yr = (bf16_t*)Y + (size_t)rowo * 768;
        yr[t] = (bf16_t)y0; yr[t + 256] = (bf16_t)y1; yr[t + 512] = (bf16_t)y2;
    } else {
        float* yr = (float*)Y + (size_t)rowo * 768;
        yr[t] = y0; yr[t + 256] = y1; yr[t + 512] = y2;
    }
}

// ---------------------------------------------------------------------------
// Helper kernels
// ---------------------------------------------------------------------------
__global__ void k_fill_u32(unsigned int* p, int n) {
    int i = blockIdx.x * 256 + threadIdx.x;
    if (i < n) p[i] = 0u;
}

// fp32 (KxN row-major) -> bf16 TRANSPOSED (NxK): dst[n*K + k] = src[k*N + n]
__global__ void k_cvt_t(const float* __restrict__ src, bf16_t* __restrict__ dst, int K, int N) {
    int idx = blockIdx.x * 256 + threadIdx.x;
    if (idx >= K * N) return;
    int n = idx / K, k = idx % K;
    dst[idx] = (bf16_t)src[(size_t)k * N + n];
}

// conv2 weight (OIHW 768x384x3x3) -> bf16 transposed [tap][n=768][k=384]
__global__ void k_cvt_conv2w(const float* __restrict__ w, bf16_t* __restrict__ wb) {
    int idx = blockIdx.x * 256 + threadIdx.x;
    if (idx >= 9 * 768 * 384) return;
    int k = idx % 384;
    int t = idx / 384;
    int n = t % 768;
    int tap = t / 768;
    int dy = tap / 3, dx = tap % 3;
    wb[idx] = (bf16_t)w[((n * 384 + k) * 3 + dy) * 3 + dx];
}

// per-tap A-row element offsets into padded conv1 activation (64 x 18 x 18 x 384)
__global__ void k_rowoff(int* __restrict__ roff) {
    int idx = blockIdx.x * 256 + threadIdx.x;
    if (idx >= 9 * 16384) return;
    int tap = idx / 16384, rowid = idx % 16384;
    int img = rowid >> 8, pos = rowid & 255;
    int y = pos >> 4, x = pos & 15;
    int dy = tap / 3, dx = tap % 3;
    roff[idx] = img * 324 * 384 + ((y + dy) * 18 + (x + dx)) * 384;
}

// conv1 3->384, 3x3 pad1, relu + 2x2 maxpool, store bf16 into padded layout
__global__ void k_conv1(const float* __restrict__ x, const float* __restrict__ w,
                        const float* __restrict__ bias, bf16_t* __restrict__ pad1) {
    int idx = blockIdx.x * 256 + threadIdx.x;
    if (idx >= 64 * 16 * 16 * 384) return;
    int oc = idx % 384;
    int t = idx / 384;
    int ox = t & 15; t >>= 4;
    int oy = t & 15; t >>= 4;
    int img = t;
    float mx = 0.0f;  // relu folded into maxpool init
    float bb = bias[oc];
    for (int dy = 0; dy < 2; ++dy)
        for (int dx = 0; dx < 2; ++dx) {
            int Y = oy * 2 + dy, X = ox * 2 + dx;
            float s = bb;
            for (int ci = 0; ci < 3; ++ci)
                for (int ky = 0; ky < 3; ++ky) {
                    int iy = Y + ky - 1;
                    if (iy < 0 || iy > 31) continue;
                    for (int kx = 0; kx < 3; ++kx) {
                        int ix = X + kx - 1;
                        if (ix < 0 || ix > 31) continue;
                        s += x[((img * 3 + ci) * 32 + iy) * 32 + ix] *
                             w[((oc * 3 + ci) * 3 + ky) * 3 + kx];
                    }
                }
            mx = fmaxf(mx, s);
        }
    pad1[(size_t)img * 324 * 384 + ((oy + 1) * 18 + (ox + 1)) * 384 + oc] = (bf16_t)mx;
}

// relu + 2x2 maxpool of conv2 GEMM output (16384x768) -> tokens into h
__global__ void k_pool(const float* __restrict__ C, float* __restrict__ h) {
    int idx = blockIdx.x * 256 + threadIdx.x;
    if (idx >= 64 * 64 * 768) return;
    int c = idx % 768;
    int t = idx / 768;
    int pos = t & 63;
    int img = t >> 6;
    int py = pos >> 3, px = pos & 7;
    float m = -3.0e38f;
    for (int dy = 0; dy < 2; ++dy)
        for (int dx = 0; dx < 2; ++dx) {
            int rr = img * 256 + (2 * py + dy) * 16 + (2 * px + dx);
            m = fmaxf(m, C[(size_t)rr * 768 + c]);
        }
    m = fmaxf(m, 0.0f);
    int b = img >> 4, il = img & 15;
    h[((size_t)b * 1025 + 1 + il * 64 + pos) * 768 + c] = m;
}

__global__ void k_cls(const float* __restrict__ cls, float* __restrict__ h) {
    int idx = blockIdx.x * 256 + threadIdx.x;
    if (idx >= 4 * 768) return;
    int b = idx / 768, c = idx % 768;
    h[(size_t)b * 1025 * 768 + c] = cls[c];
}

// h[map(r)] += tmp[r]  (dense local rows -> h rows, skipping cls)
__global__ void k_add_remap(const float* __restrict__ tmp, float* __restrict__ h) {
    int idx = blockIdx.x * 256 + threadIdx.x;
    if (idx >= 4096 * 768) return;
    int r = idx / 768, c = idx % 768;
    int hrow = (r >> 10) * 1025 + 1 + (r & 1023);
    h[(size_t)hrow * 768 + c] += tmp[idx];
}

// head: out[b, j] = lnf[cls row of b] . head_w[:, j] + head_b[j]
__global__ void k_head(const float* __restrict__ lnf, const float* __restrict__ hw,
                       const float* __restrict__ hb, float* __restrict__ out) {
    int t = threadIdx.x;
    if (t >= 40) return;
    int b = t / 10, j = t % 10;
    const float* xr = lnf + (size_t)b * 1025 * 768;
    float s = hb[j];
    for (int c = 0; c < 768; ++c) s += xr[c] * hw[c * 10 + j];
    out[t] = s;
}

// ---------------------------------------------------------------------------
// Host orchestration
// ---------------------------------------------------------------------------
static inline size_t al256(size_t x) { return (x + 255) & ~(size_t)255; }
static inline int cdiv(long long a, long long b) { return (int)((a + b - 1) / b); }

extern "C" void kernel_launch(void* const* d_in, const int* in_sizes, int n_in,
                              void* d_out, int out_size, void* d_ws, size_t ws_size,
                              hipStream_t stream) {
    (void)in_sizes; (void)n_in; (void)out_size; (void)ws_size;
    const float* x       = (const float*)d_in[0];
    const float* conv1_w = (const float*)d_in[1];
    const float* conv1_b = (const float*)d_in[2];
    const float* conv2_w = (const float*)d_in[3];
    const float* conv2_b = (const float*)d_in[4];
    const float* cls_tok = (const float*)d_in[5];
    const float* n1_w = (const float*)d_in[6],  *n1_b = (const float*)d_in[7];
    const float* gqkv_w = (const float*)d_in[8],  *gqkv_b = (const float*)d_in[9];
    const float* gproj_w = (const float*)d_in[10], *gproj_b = (const float*)d_in[11];
    const float* n2_w = (const float*)d_in[12], *n2_b = (const float*)d_in[13];
    const float* lqkv_w = (const float*)d_in[14], *lqkv_b = (const float*)d_in[15];
    const float* lproj_w = (const float*)d_in[16], *lproj_b = (const float*)d_in[17];
    const float* n3_w = (const float*)d_in[18], *n3_b = (const float*)d_in[19];
    const float* fc1_w = (const float*)d_in[20], *fc1_b = (const float*)d_in[21];
    const float* fc2_w = (const float*)d_in[22], *fc2_b = (const float*)d_in[23];
    const float* nf_w = (const float*)d_in[24], *nf_b = (const float*)d_in[25];
    const float* head_w = (const float*)d_in[26], *head_b = (const float*)d_in[27];

    // ---- workspace layout
    char* ws = (char*)d_ws;
    size_t off = 0;
    auto take = [&](size_t bytes) { size_t o = off; off = al256(off + bytes); return o; };

    const size_t NWQ = 768ull * 2304, NWP = 768ull * 768, NW1 = 768ull * 3072, NW2 = 3072ull * 768;

    float*  h     = (float*)(ws + take(4ull * 1025 * 768 * 4));
    bf16_t* lnbuf = (bf16_t*)(ws + take(4100ull * 768 * 2));
    bf16_t* attnb = (bf16_t*)(ws + take(4100ull * 768 * 2));
    float*  tmpf  = (float*)(ws + take(4100ull * 768 * 4));
    bf16_t* wq  = (bf16_t*)(ws + take(NWQ * 2));   // all weight buffers hold B^T (NxK)
    bf16_t* wp  = (bf16_t*)(ws + take(NWP * 2));
    bf16_t* wlq = (bf16_t*)(ws + take(NWQ * 2));
    bf16_t* wlp = (bf16_t*)(ws + take(NWP * 2));
    bf16_t* w1  = (bf16_t*)(ws + take(NW1 * 2));
    bf16_t* w2  = (bf16_t*)(ws + take(NW2 * 2));
    // union region: conv chain vs transformer activations (conv dead before reuse)
    size_t big0 = off;
    bf16_t* c2w   = (bf16_t*)(ws + big0);
    size_t  o1    = al256(big0 + 9ull * 768 * 384 * 2);
    bf16_t* pad1  = (bf16_t*)(ws + o1);
    size_t  o2    = al256(o1 + 64ull * 324 * 384 * 2);
    int*    roff  = (int*)(ws + o2);
    size_t  o3    = al256(o2 + 9ull * 16384 * 4);
    float*  convC = (float*)(ws + o3);
    bf16_t* qkvb  = (bf16_t*)(ws + big0);
    bf16_t* mlpb  = (bf16_t*)(ws + al256(big0 + 4100ull * 2304 * 2));

    // ---- conv stage ----
    {
        int padw = (int)(64ull * 324 * 384 * 2 / 4);
        k_fill_u32<<<cdiv(padw, 256), 256, 0, stream>>>((unsigned int*)pad1, padw);
        k_conv1<<<cdiv(64 * 16 * 16 * 384, 256), 256, 0, stream>>>(x, conv1_w, conv1_b, pad1);
        k_cvt_conv2w<<<cdiv(9 * 768 * 384, 256), 256, 0, stream>>>(conv2_w, c2w);
        k_rowoff<<<cdiv(9 * 16384, 256), 256, 0, stream>>>(roff);
        // tap 0: bias, fresh C
        k_gemm<true, true, false, false, 0, false><<<dim3(128, 3), 256, 0, stream>>>(
            pad1, 384, roff, c2w, 16384, 768, 384,
            conv2_b, nullptr, nullptr, convC, 768);
        // taps 1..8: accumulate into C
        for (int tap = 1; tap < 9; ++tap) {
            k_gemm<true, false, true, false, 0, false><<<dim3(128, 3), 256, 0, stream>>>(
                pad1, 384, roff + tap * 16384, c2w + (size_t)tap * 768 * 384,
                16384, 768, 384, nullptr, convC, nullptr, convC, 768);
        }
        k_pool<<<cdiv(64 * 64 * 768, 256), 256, 0, stream>>>(convC, h);
        k_cls<<<cdiv(4 * 768, 256), 256, 0, stream>>>(cls_tok, h);
    }

    // ---- transformer layers ----
    auto cvt_t = [&](const float* s, bf16_t* d, int K, int N) {
        k_cvt_t<<<cdiv((long long)K * N, 256), 256, 0, stream>>>(s, d, K, N);
    };
    for (int l = 0; l < 4; ++l) {
        cvt_t(gqkv_w  + (size_t)l * NWQ, wq,  768, 2304);
        cvt_t(gproj_w + (size_t)l * NWP, wp,  768, 768);
        cvt_t(lqkv_w  + (size_t)l * NWQ, wlq, 768, 2304);
        cvt_t(lproj_w + (size_t)l * NWP, wlp, 768, 768);
        cvt_t(fc1_w   + (size_t)l * NW1, w1,  768, 3072);
        cvt_t(fc2_w   + (size_t)l * NW2, w2,  3072, 768);

        // global MHA: z += proj(attn(ln1(z)))
        k_layernorm<<<4100, 256, 0, stream>>>(h, n1_w + l * 768, n1_b + l * 768, lnbuf, 1, 0);
        k_gemm<false, true, false, false, 0, true><<<dim3(33, 9), 256, 0, stream>>>(
            lnbuf, 768, nullptr, wq, 4100, 2304, 768,
            gqkv_b + (size_t)l * 2304, nullptr, nullptr, qkvb, 2304);
        k_flash<<<dim3(9, 12, 4), 256, 0, stream>>>(qkvb, 1025, 1025, attnb, 0.125f);
        k_gemm<false, true, false, true, 0, false><<<dim3(33, 3), 256, 0, stream>>>(
            attnb, 768, nullptr, wp, 4100, 768, 768,
            gproj_b + (size_t)l * 768, nullptr, h, h, 768);

        // local MHA on patch tokens: pt += proj(attn(ln2(pt)))
        k_layernorm<<<4096, 256, 0, stream>>>(h, n2_w + l * 768, n2_b + l * 768, lnbuf, 1, 1);
        k_gemm<false, true, false, false, 0, true><<<dim3(32, 9), 256, 0, stream>>>(
            lnbuf, 768, nullptr, wlq, 4096, 2304, 768,
            lqkv_b + (size_t)l * 2304, nullptr, nullptr, qkvb, 2304);
        k_flash<<<dim3(1, 12, 64), 256, 0, stream>>>(qkvb, 64, 64, attnb, 0.125f);
        k_gemm<false, true, false, false, 0, false><<<dim3(32, 3), 256, 0, stream>>>(
            attnb, 768, nullptr, wlp, 4096, 768, 768,
            lproj_b + (size_t)l * 768, nullptr, nullptr, tmpf, 768);
        k_add_remap<<<cdiv(4096 * 768, 256), 256, 0, stream>>>(tmpf, h);

        // MLP: z += fc2(gelu(fc1(ln3(z))))
        k_layernorm<<<4100, 256, 0, stream>>>(h, n3_w + l * 768, n3_b + l * 768, lnbuf, 1, 0);
        k_gemm<false, true, false, false, 1, true><<<dim3(33, 12), 256, 0, stream>>>(
            lnbuf, 768, nullptr, w1, 4100, 3072, 768,
            fc1_b + (size_t)l * 3072, nullptr, nullptr, mlpb, 3072);
        k_gemm<false, true, false, true, 0, false><<<dim3(33, 3), 256, 0, stream>>>(
            mlpb, 3072, nullptr, w2, 4100, 768, 3072,
            fc2_b + (size_t)l * 768, nullptr, h, h, 768);
    }

    // ---- final LN + head ----
    k_layernorm<<<4100, 256, 0, stream>>>(h, nf_w, nf_b, tmpf, 0, 0);
    k_head<<<1, 64, 0, stream>>>(tmpf, head_w, head_b, (float*)d_out);
}
